// Spatial_Context_Module_82454782149132
// MI455X (gfx1250) — matrix operand order repaired and checked by reading, not compiled, not run on hardware
//
#include <hip/hip_runtime.h>

// ---------------------------------------------------------------------------
// Spatial Context Module for MI455X (gfx1250, wave32, WMMA bf16 16x16x32)
// ---------------------------------------------------------------------------
typedef __attribute__((ext_vector_type(16))) __bf16        v16bf;
typedef __attribute__((ext_vector_type(8)))  float         v8f;
typedef __attribute__((ext_vector_type(8)))  unsigned int  v8u;

#define Bv   16
#define Cc   128
#define Hh   96
#define Ww   160
#define Kk   9
#define HW   (Hh * Ww)
#define RED  (Cc * Kk)            // 1152 reduction length per conv
#define NCHUNK (RED / 32)         // 36 K-chunks of 32 for 16x16x32 WMMA
#define FRAG_DW (NCHUNK * 32 * 8) // 9216 dwords of packed B frags per source
#define TILE_DW (NCHUNK * 32 * 8) // dwords of packed A per 16-row co tile
#define MAT_DW  (8 * TILE_DW)     // dwords per full 128x1152 weight matrix

__device__ __forceinline__ unsigned short f2bf(float f) {
    unsigned int u = __builtin_bit_cast(unsigned int, f);
    unsigned int r = u + 0x7FFFu + ((u >> 16) & 1u);   // round-to-nearest-even
    return (unsigned short)(r >> 16);
}

// ---------------------------------------------------------------------------
// Prep 1: repack w1[4][C][C][K] and w2[4][2][C][C][K] (fp32) into bf16 pairs
// laid out exactly as WMMA A fragments (16-bit A 16x32 layout, wave32):
//   lane 0-15 : M=lane,   K = {0..7} in v0..3, {16..23} in v4..7 (pairs)
//   lane16-31 : M=lane-16, K = {8..15} in v0..3, {24..31} in v4..7
// Aprep[mat][tile][chunk][lane][v] : one dword = two packed bf16.
// ---------------------------------------------------------------------------
__global__ void scm_prep_w(const float* __restrict__ w1,
                           const float* __restrict__ w2,
                           unsigned int* __restrict__ Aprep) {
    int t = blockIdx.x * 256 + threadIdx.x;
    const int TOT = 12 * MAT_DW;
    if (t >= TOT) return;
    int v    = t & 7;          int u = t >> 3;
    int lane = u & 31;         u >>= 5;
    int c    = u % NCHUNK;     u /= NCHUNK;
    int tile = u & 7;
    int mat  = u >> 3;

    int m    = lane & 15;
    int base = (lane >> 4) * 8;
    int k0   = (v < 4) ? (base + 2 * v) : (16 + base + 2 * (v - 4));
    int co   = tile * 16 + m;

    unsigned int pk = 0;
    for (int i = 0; i < 2; ++i) {
        int r  = c * 32 + k0 + i;
        int ci = r / Kk, kk = r % Kk;
        float val;
        if (mat < 4) {
            val = w1[(((long)mat * Cc + co) * Cc + ci) * Kk + kk];
        } else {
            int mm = mat - 4;  // mm = p*2 + j  (j=0 cur, j=1 prev)
            val = w2[(((long)mm * Cc + co) * Cc + ci) * Kk + kk];
        }
        pk |= ((unsigned int)f2bf(val)) << (16 * i);
    }
    Aprep[t] = pk;
}

// ---------------------------------------------------------------------------
// Prep 2: fold BN into scale/shift per pass: sc = g*rsqrt(v+eps), sh = b-m*sc
// bnbuf[p][which(1|2)][sel(sc|sh)][128]
// ---------------------------------------------------------------------------
__global__ void scm_prep_bn(const float* __restrict__ g1, const float* __restrict__ b1,
                            const float* __restrict__ m1, const float* __restrict__ v1,
                            const float* __restrict__ g2, const float* __restrict__ b2,
                            const float* __restrict__ m2, const float* __restrict__ v2,
                            float* __restrict__ bnbuf) {
    int t = blockIdx.x * 256 + threadIdx.x;
    if (t >= 1024) return;
    int p = t >> 8, rem = t & 255;
    int which = rem >> 7, ch = rem & 127;
    float g = (which ? g2 : g1)[p * 128 + ch];
    float b = (which ? b2 : b1)[p * 128 + ch];
    float m = (which ? m2 : m1)[p * 128 + ch];
    float v = (which ? v2 : v1)[p * 128 + ch];
    float sc = g * rsqrtf(v + 1e-5f);
    float sh = b - m * sc;
    bnbuf[((p * 2 + which) * 2 + 0) * 128 + ch] = sc;
    bnbuf[((p * 2 + which) * 2 + 1) * 128 + ch] = sh;
}

// ---------------------------------------------------------------------------
// One recurrence step: out[:, :, slice sOut] =
//   ReLU( BN( conv1d(srcCur slice, Acur) [+ conv1d(srcPrev slice, Aprev)] ) )
// Block = 256 threads = 8 waves; block owns one (b, 16-wide l-tile) N tile,
// wave w owns co rows [16w, 16w+16).
// Per source phase: cooperatively build ALL 36 B fragments (WMMA register
// order) into LDS once, then each wave streams A (prepacked, global b128)
// and B (ds_load_b128) straight into v_wmma_f32_16x16x32_bf16.
// ---------------------------------------------------------------------------
__global__ __launch_bounds__(256) void scm_step(
    const float* __restrict__ srcCur, const float* __restrict__ srcPrev,
    const unsigned int* __restrict__ Acur, const unsigned int* __restrict__ Aprev,
    const float* __restrict__ bnsc, const float* __restrict__ bnsh,
    float* __restrict__ dst,
    int sCur, int sPrev, int sOut,
    int sliceStride, int lineStride, int L) {

    __shared__ alignas(16) unsigned int  frag[FRAG_DW];   // 36 KB: B frags, one source
    __shared__ unsigned short patch[2][Cc][24];           // 12 KB: halo'd bf16 lines

    const int tid = threadIdx.x;
    const int b   = blockIdx.x / (L / 16);
    const int l0  = (blockIdx.x % (L / 16)) * 16;
    const long baseB = (long)b * Cc * HW;
    const bool hasPrev = (srcPrev != nullptr);

    // ---- stage halo'd bf16 patches for both sources ----
    for (int e = tid; e < Cc * 24; e += 256) {
        int ci = e / 24, j = e % 24;
        int l = l0 + j - 4;
        float val = 0.f;
        if (l >= 0 && l < L)
            val = srcCur[baseB + (long)ci * HW + (long)sCur * sliceStride
                         + (long)l * lineStride];
        patch[0][ci][j] = f2bf(val);
    }
    if (hasPrev) {
        for (int e = tid; e < Cc * 24; e += 256) {
            int ci = e / 24, j = e % 24;
            int l = l0 + j - 4;
            float val = 0.f;
            if (l >= 0 && l < L)
                val = srcPrev[baseB + (long)ci * HW + (long)sPrev * sliceStride
                              + (long)l * lineStride];
            patch[1][ci][j] = f2bf(val);
        }
    }
    __syncthreads();

    const int wave = tid >> 5;      // co tile 0..7
    const int lane = tid & 31;
    const int n    = lane & 15;
    const int kh   = lane >> 4;

    v8f acc = {};
    const int nSrc = hasPrev ? 2 : 1;

    for (int src = 0; src < nSrc; ++src) {
        // ---- cooperative B-fragment build (once per block per source) ----
        const unsigned short* P = &patch[src][0][0];
        for (int e = tid; e < FRAG_DW; e += 256) {
            int fv = e & 7, lp = (e >> 3) & 31, c = e >> 8;
            int fn = lp & 15, fkh = lp >> 4;
            int r0  = c * 32 + ((fv < 4) ? (fkh * 8 + 2 * fv) : (16 + fkh * 8 + 2 * (fv - 4)));
            int ci0 = r0 / Kk, kk0 = r0 - ci0 * Kk;
            int ci1 = ci0,     kk1 = kk0 + 1;
            if (kk1 == Kk) { ci1 += 1; kk1 = 0; }
            unsigned int lo = P[ci0 * 24 + kk0 + fn];
            unsigned int hi = P[ci1 * 24 + kk1 + fn];
            frag[e] = lo | (hi << 16);
        }
        __syncthreads();

        // ---- WMMA stream: A from global (prepacked), B from LDS frags ----
        const unsigned int* A = (src == 0) ? Acur : Aprev;
        const unsigned int* aBase = A + (long)wave * TILE_DW + lane * 8;
        const unsigned int* fBase = &frag[lane * 8];
        #pragma unroll 4
        for (int c = 0; c < NCHUNK; ++c) {
            v8u areg = *(const v8u*)(aBase + c * 256);
            v8u breg = *(const v8u*)(fBase + c * 256);
            acc = __builtin_amdgcn_wmma_f32_16x16x32_bf16(
                false, __builtin_bit_cast(v16bf, areg),
                false, __builtin_bit_cast(v16bf, breg),
                (short)0, acc, false, false);
        }
        __syncthreads();   // frags fully consumed before next phase rebuilds
    }

    // ---- epilogue: BN + ReLU, scatter per 16x16 f32 C/D layout ----
    const int coBase = wave * 16 + kh * 8;
    #pragma unroll
    for (int j = 0; j < 8; ++j) {
        int co = coBase + j;
        float y = acc[j] * bnsc[co] + bnsh[co];
        y = fmaxf(y, 0.f);
        dst[baseB + (long)co * HW + (long)sOut * sliceStride
            + (long)(l0 + n) * lineStride] = y;
    }
}

// ---------------------------------------------------------------------------
extern "C" void kernel_launch(void* const* d_in, const int* in_sizes, int n_in,
                              void* d_out, int out_size, void* d_ws, size_t ws_size,
                              hipStream_t stream) {
    const float* x   = (const float*)d_in[0];
    const float* w1  = (const float*)d_in[1];
    const float* w2  = (const float*)d_in[2];
    const float* g1  = (const float*)d_in[3];
    const float* b1  = (const float*)d_in[4];
    const float* m1  = (const float*)d_in[5];
    const float* v1  = (const float*)d_in[6];
    const float* g2  = (const float*)d_in[7];
    const float* b2  = (const float*)d_in[8];
    const float* m2  = (const float*)d_in[9];
    const float* v2  = (const float*)d_in[10];

    const long XELEMS = (long)Bv * Cc * Hh * Ww;         // 31,457,280 floats
    float*        inter = (float*)d_ws;                  // ping-pong tensor
    unsigned int* Aprep = (unsigned int*)(inter + XELEMS);
    float*        bnbuf = (float*)(Aprep + 12 * MAT_DW);

    scm_prep_w <<<(12 * MAT_DW + 255) / 256, 256, 0, stream>>>(w1, w2, Aprep);
    scm_prep_bn<<<4, 256, 0, stream>>>(g1, b1, m1, v1, g2, b2, m2, v2, bnbuf);

    // pass chaining: x -> ws -> d_out -> ws -> d_out
    const float* bufIn[4]  = { x, inter, (const float*)d_out, inter };
    float*       bufOut[4] = { inter, (float*)d_out, inter, (float*)d_out };
    const int VERT[4] = {1, 1, 0, 0};
    const int REV[4]  = {0, 1, 0, 1};

    for (int p = 0; p < 4; ++p) {
        const int S           = VERT[p] ? Hh : Ww;
        const int Ldim        = VERT[p] ? Ww : Hh;
        const int sliceStride = VERT[p] ? Ww : 1;
        const int lineStride  = VERT[p] ? 1  : Ww;
        const int nTiles      = Bv * Ldim / 16;

        const unsigned int* A1  = Aprep + (long)p * MAT_DW;
        const unsigned int* A2c = Aprep + (long)(4 + 2 * p + 0) * MAT_DW;
        const unsigned int* A2p = Aprep + (long)(4 + 2 * p + 1) * MAT_DW;
        const float* bn1sc = bnbuf + (p * 2 + 0) * 2 * 128;
        const float* bn1sh = bn1sc + 128;
        const float* bn2sc = bnbuf + (p * 2 + 1) * 2 * 128;
        const float* bn2sh = bn2sc + 128;

        for (int t = 0; t < S; ++t) {
            int s  = REV[p] ? (S - 1 - t) : t;
            int sp = REV[p] ? (s + 1) : (s - 1);
            if (t == 0) {
                scm_step<<<nTiles, 256, 0, stream>>>(
                    bufIn[p], nullptr, A1, nullptr, bn1sc, bn1sh,
                    bufOut[p], s, 0, s, sliceStride, lineStride, Ldim);
            } else {
                scm_step<<<nTiles, 256, 0, stream>>>(
                    bufIn[p], bufOut[p], A2c, A2p, bn2sc, bn2sh,
                    bufOut[p], s, sp, s, sliceStride, lineStride, Ldim);
            }
        }
    }
    (void)in_sizes; (void)n_in; (void)out_size; (void)ws_size;
}
